// LSTM_1279900254644
// MI455X (gfx1250) — matrix operand order repaired and checked by reading, not compiled, not run on hardware
//
#include <hip/hip_runtime.h>

// ---------------------------------------------------------------------------
// 2-layer LSTM (B=512, L=1000, H=128) persistent-workgroup kernel for gfx1250.
//
// ~201 GFLOP total, <8 MB unique global traffic -> compute/latency bound on
// the sequential t-loop. Batch rows are independent: 32 workgroups each own
// 16 rows and run all 1000 steps with weights and state resident on the WGP:
//   - W_hh1, W_ih2 as fp16 WMMA B-fragments in LDS  (2 x 128 KB, 320KB WGP LDS)
//   - W_hh2 as fp16 WMMA B-fragments in VGPRs       (128 VGPRs / wave)
//   - h1/h2 as fp16 in LDS (doubles as C-layout -> A-fragment transpose)
//   - c1/c2 + gate accumulators in f32 VGPRs
// 48 v_wmma_f32_16x16x32_f16 per wave per timestep, with A/B fragment
// prefetch + double buffering so ds_load latency overlaps the matrix pipe.
// ---------------------------------------------------------------------------

typedef __attribute__((ext_vector_type(16))) _Float16 v16h;
typedef __attribute__((ext_vector_type(8)))  _Float16 v8h;
typedef __attribute__((ext_vector_type(8)))  float    v8f;

static constexpr int BB = 512;    // batch
static constexpr int LL = 1000;   // sequence length
static constexpr int HH = 128;    // hidden
static constexpr int GG = 512;    // 4*H gates
static constexpr int ROWS = 16;   // batch rows per workgroup
static constexpr int NTHREADS = 256;  // 8 waves
static constexpr int HSTRIDE = 136;   // halves per h row (128 + 8 pad), 16B-align kept

// LDS layout (bytes)
static constexpr int OFF_W1   = 0;        // 131072  W_hh1 fp16 fragments
static constexpr int OFF_W2   = 131072;   // 131072  W_ih2 fp16 fragments
static constexpr int OFF_H1   = 262144;   // 4352    h1 fp16 [16][HSTRIDE]
static constexpr int OFF_H2   = 266496;   // 4352    h2 fp16 [16][HSTRIDE]
static constexpr int OFF_B1   = 270848;   // 2048    b_ih1+b_hh1 f32[512]
static constexpr int OFF_B2   = 272896;   // 2048    b_ih2+b_hh2 f32[512]
static constexpr int OFF_WI1  = 274944;   // 2048    W_ih1 f32[512]
static constexpr int OFF_WLIN = 276992;   // 512     W_lin f32[128]
static constexpr int OFF_X    = 277504;   // 64      x_t f32[16]
static constexpr int OFF_PART = 277568;   // 512     out partials f32[8][16]
static constexpr int SMEM_BYTES = 278080; // < 320 KB WGP LDS

__device__ __forceinline__ float sigm_f(float x) {
  return 1.0f / (1.0f + __expf(-x));
}
__device__ __forceinline__ float tanh_f(float x) {
  return 1.0f - 2.0f / (__expf(2.0f * x) + 1.0f);
}

// A-fragment (16x32 fp16) for V_WMMA_F32_16X16X32_F16 from row-major fp16 LDS.
// lanes 0-15: row=lane,    halves = K[k0+0..7],  K[k0+16..23]
// lanes16-31: row=lane-16, halves = K[k0+8..15], K[k0+24..31]
__device__ __forceinline__ v16h load_afrag(const _Float16* hbase, int lane, int k0) {
  const int row = lane & 15;
  const int off = (lane >> 4) << 3;
  const _Float16* p = hbase + row * HSTRIDE + k0 + off;
  v8h lo = *(const v8h*)(p);
  v8h hi = *(const v8h*)(p + 16);
  return __builtin_shufflevector(lo, hi, 0,1,2,3,4,5,6,7,8,9,10,11,12,13,14,15);
}

// B-fragment (32x16 fp16): lane l = N col nt*16+l%16, half h = K kt*32+8*(l/16)+(h<8?h:h+8).
// Stored pre-swizzled so each lane reads 32 contiguous bytes.
__device__ __forceinline__ v16h load_bfrag(const _Float16* wf, int kt, int nt, int lane) {
  return *(const v16h*)(wf + ((((kt << 5) + nt) << 5) + lane) * 16);
}

__device__ __forceinline__ v8f wmma16(v16h a, v16h b, v8f c) {
  return __builtin_amdgcn_wmma_f32_16x16x32_f16(false, a, false, b, (short)0, c,
                                                false, false);
}

__global__ void __launch_bounds__(NTHREADS, 1)
lstm2_wmma_kernel(const float* __restrict__ y,
                  const float* __restrict__ Wih1, const float* __restrict__ Whh1,
                  const float* __restrict__ bih1, const float* __restrict__ bhh1,
                  const float* __restrict__ Wih2, const float* __restrict__ Whh2,
                  const float* __restrict__ bih2, const float* __restrict__ bhh2,
                  const float* __restrict__ Wlin, const float* __restrict__ blin,
                  float* __restrict__ out) {
  extern __shared__ char smem[];
  _Float16* w1f   = (_Float16*)(smem + OFF_W1);
  _Float16* w2f   = (_Float16*)(smem + OFF_W2);
  _Float16* h1l   = (_Float16*)(smem + OFF_H1);
  _Float16* h2l   = (_Float16*)(smem + OFF_H2);
  float*    b1l   = (float*)(smem + OFF_B1);
  float*    b2l   = (float*)(smem + OFF_B2);
  float*    wi1l  = (float*)(smem + OFF_WI1);
  float*    wlinl = (float*)(smem + OFF_WLIN);
  float*    xbuf  = (float*)(smem + OFF_X);
  float*    part  = (float*)(smem + OFF_PART);

  const int tid  = threadIdx.x;
  const int wave = tid >> 5;
  const int lane = tid & 31;
  const int r0   = blockIdx.x * ROWS;
  const float blin0 = blin[0];

  // ---- one-time setup: fp32 -> fp16 fragment conversion into LDS -----------
  // slot s = kt*1024 + nt*32 + lane_slot l ; gate row n = nt*16 + l%16;
  // halves h -> k = kt*32 + 8*(l/16) + (h<8 ? h : h+8).
  for (int s = tid; s < 4 * 32 * 32; s += NTHREADS) {
    const int kt  = s >> 10;
    const int rem = s & 1023;
    const int nt  = rem >> 5;
    const int l   = rem & 31;
    const int k   = kt * 32 + ((l >> 4) << 3);
    _Float16* d1 = w1f + (size_t)s * 16;
    _Float16* d2 = w2f + (size_t)s * 16;
#pragma unroll
    for (int h = 0; h < 16; ++h) {
      d1[h] = (_Float16)Whh1[(nt * 16 + (l & 15)) * HH + k + (h < 8 ? h : h + 8)];
      d2[h] = (_Float16)Wih2[(nt * 16 + (l & 15)) * HH + k + (h < 8 ? h : h + 8)];
    }
  }
  for (int i = tid; i < GG; i += NTHREADS) {
    b1l[i]  = bih1[i] + bhh1[i];
    b2l[i]  = bih2[i] + bhh2[i];
    wi1l[i] = Wih1[i];
  }
  for (int i = tid; i < HH; i += NTHREADS) wlinl[i] = Wlin[i];
  for (int i = tid; i < ROWS * HSTRIDE; i += NTHREADS) {
    h1l[i] = (_Float16)0.0f;
    h2l[i] = (_Float16)0.0f;
  }
  if (tid < ROWS) xbuf[tid] = y[(size_t)(r0 + tid) * LL + 0];

  // W_hh2 fragments resident in VGPRs: this wave's 4 gate tiles x 4 K-tiles.
  v16h w3[16];
#pragma unroll
  for (int kt = 0; kt < 4; ++kt) {
#pragma unroll
    for (int j = 0; j < 4; ++j) {
      const int k  = kt * 32 + ((lane >> 4) << 3);
      const int n0 = j * HH + wave * 16;
      v16h v;
#pragma unroll
      for (int h = 0; h < 16; ++h) v[h] = (_Float16)Whh2[(n0 + (lane & 15)) * HH + k + (h < 8 ? h : h + 8)];
      w3[kt * 4 + j] = v;
    }
  }
  __syncthreads();

  // Per-wave recurrent cell state (16 rows x 16 cols, WMMA C/D layout).
  v8f c1 = {};
  v8f c2 = {};
  const int nts[4] = {wave, 8 + wave, 16 + wave, 24 + wave};
  const int col    = lane & 15;
  const int rbase  = (lane >> 4) << 3;

  // Loop-invariant per-lane constants hoisted out of the t-loop.
  float b1r[4], wir[4], b2r[4];
#pragma unroll
  for (int j = 0; j < 4; ++j) {
    const int n = j * HH + wave * 16 + col;
    b1r[j] = b1l[n];
    wir[j] = wi1l[n];
    b2r[j] = b2l[n];
  }
  float wl[16];
  if (tid < 128) {
    const int ch = tid >> 4;
#pragma unroll
    for (int j2 = 0; j2 < 16; ++j2) wl[j2] = wlinl[ch * 16 + j2];
  }

  for (int t = 0; t < LL; ++t) {
    // ---- layer 1 gates: b1 + x_t (x) W_ih1  +  h1(t-1) @ W_hh1^T ----------
    const float4 xa = *(const float4*)(xbuf + rbase);
    const float4 xb = *(const float4*)(xbuf + rbase + 4);
    const float xr[8] = {xa.x, xa.y, xa.z, xa.w, xb.x, xb.y, xb.z, xb.w};
    v8f acc[4];
#pragma unroll
    for (int j = 0; j < 4; ++j) {
      v8f a;
#pragma unroll
      for (int r = 0; r < 8; ++r) a[r] = b1r[j] + xr[r] * wir[j];
      acc[j] = a;
    }
    // preload all A-fragments of h1(t-1); double-buffer the B-fragments
    v16h ah[4];
#pragma unroll
    for (int kt = 0; kt < 4; ++kt) ah[kt] = load_afrag(h1l, lane, kt * 32);
    v16h bcur[4];
#pragma unroll
    for (int j = 0; j < 4; ++j) bcur[j] = load_bfrag(w1f, 0, nts[j], lane);
#pragma unroll
    for (int kt = 0; kt < 4; ++kt) {
      v16h bnext[4];
      if (kt < 3) {
#pragma unroll
        for (int j = 0; j < 4; ++j) bnext[j] = load_bfrag(w1f, kt + 1, nts[j], lane);
      }
#pragma unroll
      for (int j = 0; j < 4; ++j) acc[j] = wmma16(ah[kt], bcur[j], acc[j]);
      if (kt < 3) {
#pragma unroll
        for (int j = 0; j < 4; ++j) bcur[j] = bnext[j];
      }
    }
    __syncthreads();  // S1: all reads of h1(t-1) and xbuf(t) complete

    // stage x(t+1); emit out(t-1) from previous step's partials
    if (t + 1 < LL && tid < ROWS)
      xbuf[tid] = y[(size_t)(r0 + tid) * LL + (t + 1)];
    if (tid < ROWS && t + 8 < LL)
      __builtin_prefetch(&y[(size_t)(r0 + tid) * LL + t + 8], 0, 0);
    if (t > 0 && wave == 1 && lane < 16) {
      float s = blin0;
#pragma unroll
      for (int w = 0; w < 8; ++w) s += part[w * 16 + lane];
      out[(size_t)(r0 + lane) * LL + (t - 1)] = s;
    }

    // layer-1 nonlinearities; write h1(t) fp16 into LDS
#pragma unroll
    for (int r = 0; r < 8; ++r) {
      const float ig = sigm_f(acc[0][r]);
      const float fg = sigm_f(acc[1][r]);
      const float gg = tanh_f(acc[2][r]);
      const float og = sigm_f(acc[3][r]);
      const float c  = fg * c1[r] + ig * gg;
      c1[r] = c;
      h1l[(rbase + r) * HSTRIDE + wave * 16 + col] = (_Float16)(og * tanh_f(c));
    }
    __syncthreads();  // S2: h1(t) visible to all waves

    // ---- layer 2 gates: b2 + h1(t) @ W_ih2^T + h2(t-1) @ W_hh2^T ----------
#pragma unroll
    for (int j = 0; j < 4; ++j) {
      v8f a;
#pragma unroll
      for (int r = 0; r < 8; ++r) a[r] = b2r[j];
      acc[j] = a;
    }
    v16h a1h[4], a2h[4];
#pragma unroll
    for (int kt = 0; kt < 4; ++kt) {
      a1h[kt] = load_afrag(h1l, lane, kt * 32);
      a2h[kt] = load_afrag(h2l, lane, kt * 32);
    }
#pragma unroll
    for (int kt = 0; kt < 4; ++kt) {
      // issue the W_ih2 fragment loads ...
      v16h b2f[4];
#pragma unroll
      for (int j = 0; j < 4; ++j) b2f[j] = load_bfrag(w2f, kt, nts[j], lane);
      // ... and hide them behind the register-resident W_hh2 WMMAs
#pragma unroll
      for (int j = 0; j < 4; ++j) acc[j] = wmma16(a2h[kt], w3[kt * 4 + j], acc[j]);
#pragma unroll
      for (int j = 0; j < 4; ++j) acc[j] = wmma16(a1h[kt], b2f[j], acc[j]);
    }
    __syncthreads();  // S3: all reads of h2(t-1) complete

#pragma unroll
    for (int r = 0; r < 8; ++r) {
      const float ig = sigm_f(acc[0][r]);
      const float fg = sigm_f(acc[1][r]);
      const float gg = tanh_f(acc[2][r]);
      const float og = sigm_f(acc[3][r]);
      const float c  = fg * c2[r] + ig * gg;
      c2[r] = c;
      h2l[(rbase + r) * HSTRIDE + wave * 16 + col] = (_Float16)(og * tanh_f(c));
    }
    __syncthreads();  // S4: h2(t) visible

    // output projection partials: thread (chunk,row) reduces 16 columns
    if (tid < 128) {
      const int row = tid & 15;
      const int ch  = tid >> 4;
      v16h hv = *(const v16h*)(h2l + row * HSTRIDE + ch * 16);
      float s = 0.0f;
#pragma unroll
      for (int j2 = 0; j2 < 16; ++j2) s += (float)hv[j2] * wl[j2];
      part[ch * 16 + row] = s;
    }
  }

  __syncthreads();
  // flush out(t = L-1)
  if (wave == 1 && lane < 16) {
    float s = blin0;
#pragma unroll
    for (int w = 0; w < 8; ++w) s += part[w * 16 + lane];
    out[(size_t)(r0 + lane) * LL + (LL - 1)] = s;
  }
}

extern "C" void kernel_launch(void* const* d_in, const int* in_sizes, int n_in,
                              void* d_out, int out_size, void* d_ws, size_t ws_size,
                              hipStream_t stream) {
  (void)in_sizes; (void)n_in; (void)out_size; (void)d_ws; (void)ws_size;
  const float* y    = (const float*)d_in[0];
  const float* Wih1 = (const float*)d_in[1];
  const float* Whh1 = (const float*)d_in[2];
  const float* bih1 = (const float*)d_in[3];
  const float* bhh1 = (const float*)d_in[4];
  const float* Wih2 = (const float*)d_in[5];
  const float* Whh2 = (const float*)d_in[6];
  const float* bih2 = (const float*)d_in[7];
  const float* bhh2 = (const float*)d_in[8];
  const float* Wlin = (const float*)d_in[9];
  const float* blin = (const float*)d_in[10];
  float* out = (float*)d_out;

  dim3 grid(BB / ROWS);   // 32 independent 16-row LSTM slices
  dim3 block(NTHREADS);   // 8 waves
  hipLaunchKernelGGL(lstm2_wmma_kernel, grid, block, SMEM_BYTES, stream,
                     y, Wih1, Whh1, bih1, bhh1, Wih2, Whh2, bih2, bhh2,
                     Wlin, blin, out);
}
